// Attention_50886772523162
// MI455X (gfx1250) — matrix operand order repaired and checked by reading, not compiled, not run on hardware
//
#include <hip/hip_runtime.h>
#include <hip/hip_bf16.h>

// Problem constants (reference: b=2, n=2048, d=1024, h=16, hd=64)
#define B_   2
#define N_   2048
#define D_   1024
#define H_   16
#define HD_  64
#define BN_  (B_ * N_)            // 4096 rows
#define SCALE_ 0.125f             // hd^-0.5
#define MASKV_ (-50000.0f)

typedef __attribute__((ext_vector_type(8)))  unsigned short ush8;
typedef __attribute__((ext_vector_type(16))) __bf16        v16bf;
typedef __attribute__((ext_vector_type(8)))  float         v8f;

union FragU { ush8 h[2]; v16bf v; };

__device__ __forceinline__ unsigned short f2bf(float f) {
  unsigned int u = __builtin_bit_cast(unsigned int, f);
  u += 0x7FFFu + ((u >> 16) & 1u);           // round-to-nearest-even
  return (unsigned short)(u >> 16);
}

// A fragment (16x32 bf16, M = lane%16): chunks K = {k0..k0+7, k0+16..k0+23}, k0 = lane<16?0:8
__device__ __forceinline__ v16bf load_a(const unsigned short* base, int ld, int lane) {
  int m  = lane & 15;
  int k0 = (lane < 16) ? 0 : 8;
  const unsigned short* p = base + (size_t)m * ld + k0;
  FragU u;
  u.h[0] = *(const ush8*)(p);
  u.h[1] = *(const ush8*)(p + 16);
  return u.v;
}

// B fragment (32x16 bf16, N = lane%16, storage is N-major with stride ld along K):
// chunks K = {k0..k0+7, k0+16..k0+23}, k0 = lane<16?0:8
__device__ __forceinline__ v16bf load_b(const unsigned short* base, int ld, int lane) {
  int nn = lane & 15;
  int k0 = (lane < 16) ? 0 : 8;
  const unsigned short* p = base + (size_t)nn * ld + k0;
  FragU u;
  u.h[0] = *(const ush8*)(p);
  u.h[1] = *(const ush8*)(p + 16);
  return u.v;
}

#define WMMA_BF16(A, Bf, C) \
  __builtin_amdgcn_wmma_f32_16x16x32_bf16(false, (A), false, (Bf), (short)0, (C), false, false)

// generic pointer to __shared__ -> 32-bit LDS byte address (aperture: low 32 bits)
__device__ __forceinline__ unsigned lds_addr(const void* p) {
  return (unsigned)(uintptr_t)p;
}

// ---------------- conversion kernels ----------------

__global__ void k_cvt(const float* __restrict__ src, unsigned short* __restrict__ dst, int n) {
  int i = blockIdx.x * blockDim.x + threadIdx.x;
  if (i < n) dst[i] = f2bf(src[i]);
}

// W: rows x cols (row-major)  ->  WT: cols x rows (row-major), i.e. N-major, contiguous along K
__global__ void k_transpose_cvt(const float* __restrict__ W, unsigned short* __restrict__ WT,
                                int rows, int cols) {
  int idx = blockIdx.x * blockDim.x + threadIdx.x;
  if (idx >= rows * cols) return;
  int i = idx % rows;     // input row (= K index)
  int o = idx / rows;     // input col (= N index)
  WT[(size_t)o * rows + i] = f2bf(W[(size_t)i * cols + o]);
}

// ---------------- fused QKV projection ----------------
// C[4096 x 3072] = xb[4096 x 1024] * [WqT ; WkvT]^T, one wave = 32 rows x 64 cols,
// register-pipelined K loop (prefetch next K-step while 8 WMMAs run).
__global__ __launch_bounds__(128) void k_proj(const unsigned short* __restrict__ xb,
                                              const unsigned short* __restrict__ WqT,
                                              const unsigned short* __restrict__ WkvT,
                                              unsigned short* __restrict__ Qh,
                                              unsigned short* __restrict__ Kh,
                                              unsigned short* __restrict__ Vt) {
  const int lane  = threadIdx.x & 31;
  const int wid   = blockIdx.x * (blockDim.x >> 5) + (threadIdx.x >> 5);
  const int NSTR  = 48;                       // 3072 / 64 output stripes
  const int tm    = wid / NSTR;               // row tile [0,128)
  const int s     = wid % NSTR;               // stripe
  const int m0    = tm * 32;

  const unsigned short* Bbase =
      (s < 16) ? (WqT  + (size_t)s        * 64 * D_)
               : (WkvT + (size_t)(s - 16) * 64 * D_);
  const unsigned short* A0 = xb + (size_t)m0 * D_;
  const unsigned short* A1 = A0 + (size_t)16 * D_;

  v8f acc[2][4];
#pragma unroll
  for (int mt = 0; mt < 2; ++mt)
#pragma unroll
    for (int t = 0; t < 4; ++t) acc[mt][t] = v8f{};

  v16bf a0 = load_a(A0, D_, lane);
  v16bf a1 = load_a(A1, D_, lane);
  v16bf b0 = load_b(Bbase + (size_t)0  * D_, D_, lane);
  v16bf b1 = load_b(Bbase + (size_t)16 * D_, D_, lane);
  v16bf b2 = load_b(Bbase + (size_t)32 * D_, D_, lane);
  v16bf b3 = load_b(Bbase + (size_t)48 * D_, D_, lane);

  for (int kk = 0; kk < D_ - 32; kk += 32) {
    int nk = kk + 32;
    v16bf na0 = load_a(A0 + nk, D_, lane);
    v16bf na1 = load_a(A1 + nk, D_, lane);
    v16bf nb0 = load_b(Bbase + (size_t)0  * D_ + nk, D_, lane);
    v16bf nb1 = load_b(Bbase + (size_t)16 * D_ + nk, D_, lane);
    v16bf nb2 = load_b(Bbase + (size_t)32 * D_ + nk, D_, lane);
    v16bf nb3 = load_b(Bbase + (size_t)48 * D_ + nk, D_, lane);
    acc[0][0] = WMMA_BF16(a0, b0, acc[0][0]);
    acc[0][1] = WMMA_BF16(a0, b1, acc[0][1]);
    acc[0][2] = WMMA_BF16(a0, b2, acc[0][2]);
    acc[0][3] = WMMA_BF16(a0, b3, acc[0][3]);
    acc[1][0] = WMMA_BF16(a1, b0, acc[1][0]);
    acc[1][1] = WMMA_BF16(a1, b1, acc[1][1]);
    acc[1][2] = WMMA_BF16(a1, b2, acc[1][2]);
    acc[1][3] = WMMA_BF16(a1, b3, acc[1][3]);
    a0 = na0; a1 = na1; b0 = nb0; b1 = nb1; b2 = nb2; b3 = nb3;
  }
  acc[0][0] = WMMA_BF16(a0, b0, acc[0][0]);
  acc[0][1] = WMMA_BF16(a0, b1, acc[0][1]);
  acc[0][2] = WMMA_BF16(a0, b2, acc[0][2]);
  acc[0][3] = WMMA_BF16(a0, b3, acc[0][3]);
  acc[1][0] = WMMA_BF16(a1, b0, acc[1][0]);
  acc[1][1] = WMMA_BF16(a1, b1, acc[1][1]);
  acc[1][2] = WMMA_BF16(a1, b2, acc[1][2]);
  acc[1][3] = WMMA_BF16(a1, b3, acc[1][3]);

  // destination routing
  unsigned short* dst;
  int h;
  bool isV = false;
  if (s < 16)      { dst = Qh; h = s; }
  else if (s < 32) { dst = Kh; h = s - 16; }
  else             { dst = Vt; h = s - 32; isV = true; }

  const int hiOff = (lane >> 4) << 3;         // 0 or 8
  const int cl    = lane & 15;
#pragma unroll
  for (int mt = 0; mt < 2; ++mt)
#pragma unroll
    for (int t = 0; t < 4; ++t)
#pragma unroll
      for (int e = 0; e < 8; ++e) {
        int m   = m0 + mt * 16 + e + hiOff;
        int bi  = m >> 11;                    // /2048
        int i   = m & (N_ - 1);
        int col = t * 16 + cl;
        size_t idx = isV
            ? ((size_t)(bi * H_ + h) * HD_ + col) * N_ + i
            : ((size_t)(bi * H_ + h) * N_ + i) * HD_ + col;
        dst[idx] = f2bf(acc[mt][t][e]);
      }
}

// ---------------- flash attention ----------------
// one wave = one 16-row i-tile of one (b,h); online softmax; causal.
// K/V j-blocks are double-buffered in LDS via async-to-LDS copies (ASYNCcnt).

__device__ __forceinline__ void stage_kv_async(const unsigned short* Kbase,
                                               const unsigned short* Vbase,
                                               int j0, unsigned klds, unsigned vlds,
                                               int lane) {
  // K block: contiguous 4KB (32 rows x 64 bf16) at Kbase + j0*64
  const char* kg = (const char*)(Kbase + (size_t)j0 * HD_);
#pragma unroll
  for (int c = 0; c < 8; ++c) {
    unsigned ld = klds + (unsigned)(c * 32 + lane) * 16;
    unsigned long long g = (unsigned long long)(uintptr_t)(kg + (c * 32 + lane) * 16);
    asm volatile("global_load_async_to_lds_b128 %0, %1, off" :: "v"(ld), "v"(g) : "memory");
  }
  // V block: 64 rows (hd) x 64B each at Vbase + row*N_ + j0  -> LDS rows of 32 elems
#pragma unroll
  for (int c = 0; c < 8; ++c) {
    int chunk = c * 32 + lane;                 // 0..255
    int row   = chunk >> 2;
    int off   = (chunk & 3) * 16;              // bytes within 64B row
    unsigned ld = vlds + (unsigned)(row * 64 + off);
    unsigned long long g =
        (unsigned long long)(uintptr_t)((const char*)(Vbase + (size_t)row * N_ + j0) + off);
    asm volatile("global_load_async_to_lds_b128 %0, %1, off" :: "v"(ld), "v"(g) : "memory");
  }
}

__global__ __launch_bounds__(128) void k_attn(const unsigned short* __restrict__ Qh,
                                              const unsigned short* __restrict__ Kh,
                                              const unsigned short* __restrict__ Vt,
                                              unsigned short* __restrict__ O) {
  __shared__ __align__(16) unsigned short kbuf[4][2][32 * HD_];   // 4KB per buffer
  __shared__ __align__(16) unsigned short vbuf[4][2][HD_ * 32];   // 4KB per buffer
  __shared__ __align__(16) unsigned short pbuf[4][16 * 32];

  const int lane  = threadIdx.x & 31;
  const int wslot = (threadIdx.x >> 5) & 3;
  const int wid   = blockIdx.x * (blockDim.x >> 5) + (threadIdx.x >> 5);
  const int NT    = N_ / 16;                  // 128 i-tiles
  const int bh    = wid / NT;                 // [0, 32)
  const int it    = wid % NT;
  const int i0    = it * 16;
  const int bi    = bh >> 4;
  const int h     = bh & 15;

  const unsigned short* Qbase = Qh + ((size_t)bh * N_ + i0) * HD_;
  const unsigned short* Kbase = Kh + (size_t)bh * N_ * HD_;
  const unsigned short* Vbase = Vt + (size_t)bh * HD_ * N_;

  const v16bf qa0 = load_a(Qbase,      HD_, lane);   // K-dim hd 0..31
  const v16bf qa1 = load_a(Qbase + 32, HD_, lane);   // K-dim hd 32..63

  const int hiOff = (lane >> 4) << 3;
  const int cl    = lane & 15;

  v8f acc[4] = {v8f{}, v8f{}, v8f{}, v8f{}};
  float m_run[8], l_run[8];
#pragma unroll
  for (int e = 0; e < 8; ++e) { m_run[e] = -3.0e38f; l_run[e] = 0.0f; }

  unsigned short* pb = &pbuf[wslot][0];
  const unsigned kl[2] = { lds_addr(&kbuf[wslot][0][0]), lds_addr(&kbuf[wslot][1][0]) };
  const unsigned vl[2] = { lds_addr(&vbuf[wslot][0][0]), lds_addr(&vbuf[wslot][1][0]) };

  const int jmax = i0 + 15;
  stage_kv_async(Kbase, Vbase, 0, kl[0], vl[0], lane);

  int pp = 0;
  for (int j0 = 0; j0 <= jmax; j0 += 32) {
    const bool hasNext = (j0 + 32) <= jmax;
    if (hasNext) {
      stage_kv_async(Kbase, Vbase, j0 + 32, kl[pp ^ 1], vl[pp ^ 1], lane);
      asm volatile("s_wait_asynccnt 0x10" ::: "memory");  // in-order: current 16 done
    } else {
      asm volatile("s_wait_asynccnt 0x0" ::: "memory");
    }

    const unsigned short* kc = &kbuf[wslot][pp][0];
    const unsigned short* vc = &vbuf[wslot][pp][0];
    v16bf kb0 = load_b(kc,                HD_, lane);  // rows j0..j0+15, hd 0..31
    v16bf kb1 = load_b(kc + 32,           HD_, lane);  // rows j0..j0+15, hd 32..63
    v16bf kb2 = load_b(kc + 16 * HD_,     HD_, lane);  // rows j0+16..j0+31
    v16bf kb3 = load_b(kc + 16 * HD_ + 32, HD_, lane);
    v16bf vb0 = load_b(vc + 0 * 16 * 32, 32, lane);    // V cols 0..15
    v16bf vb1 = load_b(vc + 1 * 16 * 32, 32, lane);
    v16bf vb2 = load_b(vc + 2 * 16 * 32, 32, lane);
    v16bf vb3 = load_b(vc + 3 * 16 * 32, 32, lane);

    v8f s0 = {}, s1 = {};
    s0 = WMMA_BF16(qa0, kb0, s0);
    s0 = WMMA_BF16(qa1, kb1, s0);
    s1 = WMMA_BF16(qa0, kb2, s1);
    s1 = WMMA_BF16(qa1, kb3, s1);

    // scale + causal mask + block row-max
    float bm[8];
#pragma unroll
    for (int e = 0; e < 8; ++e) {
      int row = i0 + e + hiOff;
      int c0  = j0 + cl;
      int c1  = c0 + 16;
      float a = (c0 > row) ? MASKV_ : s0[e] * SCALE_;
      float b = (c1 > row) ? MASKV_ : s1[e] * SCALE_;
      s0[e] = a; s1[e] = b;
      bm[e] = fmaxf(a, b);
    }
#pragma unroll
    for (int e = 0; e < 8; ++e)
      for (int off = 1; off < 16; off <<= 1)
        bm[e] = fmaxf(bm[e], __shfl_xor(bm[e], off, 32));

    // online softmax update; P -> LDS (bf16)
    float rs[8], corr[8];
#pragma unroll
    for (int e = 0; e < 8; ++e) {
      float mn = fmaxf(m_run[e], bm[e]);
      corr[e]  = __expf(m_run[e] - mn);
      m_run[e] = mn;
      float p0 = __expf(s0[e] - mn);
      float p1 = __expf(s1[e] - mn);
      int row = e + hiOff;
      pb[row * 32 + cl]      = f2bf(p0);
      pb[row * 32 + 16 + cl] = f2bf(p1);
      rs[e] = p0 + p1;
    }
#pragma unroll
    for (int e = 0; e < 8; ++e)
      for (int off = 1; off < 16; off <<= 1)
        rs[e] += __shfl_xor(rs[e], off, 32);
#pragma unroll
    for (int e = 0; e < 8; ++e)
      l_run[e] = l_run[e] * corr[e] + rs[e];
#pragma unroll
    for (int t = 0; t < 4; ++t)
#pragma unroll
      for (int e = 0; e < 8; ++e)
        acc[t][e] *= corr[e];

    asm volatile("s_wait_dscnt 0" ::: "memory");     // P stores visible to same-wave loads
    v16bf pa = load_a(pb, 32, lane);                 // P as A fragment (16x32)

    acc[0] = WMMA_BF16(pa, vb0, acc[0]);
    acc[1] = WMMA_BF16(pa, vb1, acc[1]);
    acc[2] = WMMA_BF16(pa, vb2, acc[2]);
    acc[3] = WMMA_BF16(pa, vb3, acc[3]);

    asm volatile("s_wait_dscnt 0" ::: "memory");     // P loads done before next overwrite
    pp ^= 1;
  }

  // normalize and store O (merged heads, row-major [b*n][d], bf16)
#pragma unroll
  for (int t = 0; t < 4; ++t) {
#pragma unroll
    for (int e = 0; e < 8; ++e) {
      float o = acc[t][e] / l_run[e];
      int row = i0 + e + hiOff;
      int col = h * HD_ + t * 16 + cl;
      O[((size_t)bi * N_ + row) * D_ + col] = f2bf(o);
    }
  }
}

// ---------------- output projection ----------------
// out[4096 x 1024] (fp32) = O_bf16[4096 x 1024] * Wo ; WoT is N-major.
// one wave = 32 rows x 64 cols, register-pipelined like k_proj.
__global__ __launch_bounds__(128) void k_out(const unsigned short* __restrict__ O,
                                             const unsigned short* __restrict__ WoT,
                                             float* __restrict__ out) {
  const int lane = threadIdx.x & 31;
  const int wid  = blockIdx.x * (blockDim.x >> 5) + (threadIdx.x >> 5);
  const int NSTR = 16;                        // 1024 / 64
  const int tm   = wid / NSTR;                // [0,128)
  const int s    = wid % NSTR;
  const int m0   = tm * 32;

  const unsigned short* A0 = O + (size_t)m0 * D_;
  const unsigned short* A1 = A0 + (size_t)16 * D_;
  const unsigned short* Bbase = WoT + (size_t)s * 64 * D_;

  v8f acc[2][4];
#pragma unroll
  for (int mt = 0; mt < 2; ++mt)
#pragma unroll
    for (int t = 0; t < 4; ++t) acc[mt][t] = v8f{};

  v16bf a0 = load_a(A0, D_, lane);
  v16bf a1 = load_a(A1, D_, lane);
  v16bf b0 = load_b(Bbase + (size_t)0  * D_, D_, lane);
  v16bf b1 = load_b(Bbase + (size_t)16 * D_, D_, lane);
  v16bf b2 = load_b(Bbase + (size_t)32 * D_, D_, lane);
  v16bf b3 = load_b(Bbase + (size_t)48 * D_, D_, lane);

  for (int kk = 0; kk < D_ - 32; kk += 32) {
    int nk = kk + 32;
    v16bf na0 = load_a(A0 + nk, D_, lane);
    v16bf na1 = load_a(A1 + nk, D_, lane);
    v16bf nb0 = load_b(Bbase + (size_t)0  * D_ + nk, D_, lane);
    v16bf nb1 = load_b(Bbase + (size_t)16 * D_ + nk, D_, lane);
    v16bf nb2 = load_b(Bbase + (size_t)32 * D_ + nk, D_, lane);
    v16bf nb3 = load_b(Bbase + (size_t)48 * D_ + nk, D_, lane);
    acc[0][0] = WMMA_BF16(a0, b0, acc[0][0]);
    acc[0][1] = WMMA_BF16(a0, b1, acc[0][1]);
    acc[0][2] = WMMA_BF16(a0, b2, acc[0][2]);
    acc[0][3] = WMMA_BF16(a0, b3, acc[0][3]);
    acc[1][0] = WMMA_BF16(a1, b0, acc[1][0]);
    acc[1][1] = WMMA_BF16(a1, b1, acc[1][1]);
    acc[1][2] = WMMA_BF16(a1, b2, acc[1][2]);
    acc[1][3] = WMMA_BF16(a1, b3, acc[1][3]);
    a0 = na0; a1 = na1; b0 = nb0; b1 = nb1; b2 = nb2; b3 = nb3;
  }
  acc[0][0] = WMMA_BF16(a0, b0, acc[0][0]);
  acc[0][1] = WMMA_BF16(a0, b1, acc[0][1]);
  acc[0][2] = WMMA_BF16(a0, b2, acc[0][2]);
  acc[0][3] = WMMA_BF16(a0, b3, acc[0][3]);
  acc[1][0] = WMMA_BF16(a1, b0, acc[1][0]);
  acc[1][1] = WMMA_BF16(a1, b1, acc[1][1]);
  acc[1][2] = WMMA_BF16(a1, b2, acc[1][2]);
  acc[1][3] = WMMA_BF16(a1, b3, acc[1][3]);

  const int hiOff = (lane >> 4) << 3;
  const int cl    = lane & 15;
#pragma unroll
  for (int mt = 0; mt < 2; ++mt)
#pragma unroll
    for (int t = 0; t < 4; ++t)
#pragma unroll
      for (int e = 0; e < 8; ++e) {
        int m   = m0 + mt * 16 + e + hiOff;
        int col = s * 64 + t * 16 + cl;
        out[(size_t)m * D_ + col] = acc[mt][t][e];
      }
}

// ---------------- host side ----------------

extern "C" void kernel_launch(void* const* d_in, const int* in_sizes, int n_in,
                              void* d_out, int out_size, void* d_ws, size_t ws_size,
                              hipStream_t stream) {
  const float* x   = (const float*)d_in[0];   // [2, 2048, 1024]
  const float* Wq  = (const float*)d_in[1];   // [1024, 1024]
  const float* Wkv = (const float*)d_in[2];   // [1024, 2048]
  const float* Wo  = (const float*)d_in[3];   // [1024, 1024]

  unsigned short* xb   = (unsigned short*)d_ws;                 // 4096*1024
  unsigned short* WqT  = xb   + (size_t)BN_ * D_;               // 1024*1024
  unsigned short* WkvT = WqT  + (size_t)D_ * D_;                // 2048*1024
  unsigned short* WoT  = WkvT + (size_t)2 * D_ * D_;            // 1024*1024
  unsigned short* Qh   = WoT  + (size_t)D_ * D_;                // 4096*1024
  unsigned short* Kh   = Qh   + (size_t)BN_ * D_;
  unsigned short* Vt   = Kh   + (size_t)BN_ * D_;
  unsigned short* O    = Vt   + (size_t)BN_ * D_;

  // 1) convert / transpose to bf16
  k_cvt<<<(BN_ * D_ + 255) / 256, 256, 0, stream>>>(x, xb, BN_ * D_);
  k_transpose_cvt<<<(D_ * D_ + 255) / 256, 256, 0, stream>>>(Wq, WqT, D_, D_);
  k_transpose_cvt<<<(D_ * 2 * D_ + 255) / 256, 256, 0, stream>>>(Wkv, WkvT, D_, 2 * D_);
  k_transpose_cvt<<<(D_ * D_ + 255) / 256, 256, 0, stream>>>(Wo, WoT, D_, D_);

  // 2) fused QKV projection: 128 row-tiles * 48 stripes = 6144 waves / 4 per block
  k_proj<<<1536, 128, 0, stream>>>(xb, WqT, WkvT, Qh, Kh, Vt);

  // 3) attention: 2*16*128 = 4096 waves / 4 per block
  k_attn<<<1024, 128, 0, stream>>>(Qh, Kh, Vt, O);

  // 4) output projection: 128 * 16 = 2048 waves / 4 per block
  k_out<<<512, 128, 0, stream>>>(O, WoT, (float*)d_out);
}
